// QJLCompressor_19490561589515
// MI455X (gfx1250) — matrix operand order repaired and checked by reading, not compiled, not run on hardware
//
#include <hip/hip_runtime.h>

// ---------------------------------------------------------------------------
// QJL score correction, fused as one bf16 WMMA GEMM per head:
//   out[h] = [q | qG/R] (N x 192)  @  [k_quant | sign((k_orig-k_quant)G)]^T (192 x T)
// B=1, H=32, N=T=2048, D=128, R=64  ->  K = D+R = 192
//
// Roofline: output = 512 MB f32 (written once)  ->  ~22 us at 23.3 TB/s.
// Packed bf16 panels (50 MB) fit in the 192 MB L2, so panel re-reads across
// tiles are L2 hits as long as the streaming output stores are non-temporal.
// ---------------------------------------------------------------------------

typedef __attribute__((ext_vector_type(16))) __bf16        v16bf;
typedef __attribute__((ext_vector_type(8)))  float         v8f;
typedef __attribute__((ext_vector_type(4)))  unsigned int  uint4v;

#define HEADS 32
#define NROWS 2048
#define TCOLS 2048
#define DDIM  128
#define RANK  64
#define KTOT  192   // D + R
#define TILE  128   // output tile is 128x128
#define KCH   64    // K staged through LDS in chunks of 64 (2 wmma K-steps)

__device__ __forceinline__ unsigned short f2bf(float x) {
  unsigned int u = __builtin_bit_cast(unsigned int, x);
  unsigned int r = u + 0x7FFFu + ((u >> 16) & 1u);   // round-to-nearest-even
  return (unsigned short)(r >> 16);
}

// ---------------------------------------------------------------------------
// Prep A panel: Ap[h*N + n][0:128] = bf16(q), Ap[..][128+r] = bf16(q.G[:,r]/R)
// One 64-thread block (2 waves) per (h,n) row. G (32KB) stays resident in L2.
// q is a single-use stream -> non-temporal loads.
// ---------------------------------------------------------------------------
__global__ __launch_bounds__(64)
void qjl_prep_q(const float* __restrict__ q, const float* __restrict__ G,
                unsigned short* __restrict__ Ap) {
  __shared__ __attribute__((aligned(16))) float row[DDIM];
  const size_t r0 = (size_t)blockIdx.x;              // h*N + n
  const float* qr = q + r0 * DDIM;
  unsigned short* ar = Ap + r0 * KTOT;
  const int tid = threadIdx.x;

  float q0 = __builtin_nontemporal_load(&qr[tid]);
  float q1 = __builtin_nontemporal_load(&qr[tid + 64]);
  row[tid]      = q0;
  row[tid + 64] = q1;
  ar[tid]       = f2bf(q0);
  ar[tid + 64]  = f2bf(q1);
  __syncthreads();

  float acc = 0.f;                                   // projection r = tid
#pragma unroll 8
  for (int d = 0; d < DDIM; ++d) acc += row[d] * G[d * RANK + tid];
  ar[DDIM + tid] = f2bf(acc * (1.0f / (float)RANK)); // alpha^2 = 1/R folded here
}

// ---------------------------------------------------------------------------
// Prep B panel: Bp[h*T + t][0:128] = bf16(k_quant),
//               Bp[..][128+r] = bf16(sign((k_orig-k_quant).G[:,r]))
// k_orig / k_quant are single-use streams -> non-temporal loads.
// ---------------------------------------------------------------------------
__global__ __launch_bounds__(64)
void qjl_prep_k(const float* __restrict__ ko, const float* __restrict__ kq,
                const float* __restrict__ G, unsigned short* __restrict__ Bp) {
  __shared__ __attribute__((aligned(16))) float e[DDIM];
  const size_t r0 = (size_t)blockIdx.x;              // h*T + t
  const float* kor = ko + r0 * DDIM;
  const float* kqr = kq + r0 * DDIM;
  unsigned short* br = Bp + r0 * KTOT;
  const int tid = threadIdx.x;

  float kq0 = __builtin_nontemporal_load(&kqr[tid]);
  float kq1 = __builtin_nontemporal_load(&kqr[tid + 64]);
  e[tid]       = __builtin_nontemporal_load(&kor[tid])      - kq0;
  e[tid + 64]  = __builtin_nontemporal_load(&kor[tid + 64]) - kq1;
  br[tid]      = f2bf(kq0);
  br[tid + 64] = f2bf(kq1);
  __syncthreads();

  float acc = 0.f;
#pragma unroll 8
  for (int d = 0; d < DDIM; ++d) acc += e[d] * G[d * RANK + tid];
  float s = (acc > 0.f) ? 1.f : ((acc < 0.f) ? -1.f : 0.f);
  br[DDIM + tid] = f2bf(s);
}

// ---------------------------------------------------------------------------
// Batched GEMM: out[h] = A[h] (2048x192) x B[h]^T (192x2048), f32 accumulate.
// 256 threads = 8 wave32s arranged 4(M) x 2(N); each wave owns a 32x64 block
// = 2x4 WMMA 16x16 accumulators. K staged in 128x64 bf16 LDS panels (32 KB).
// Output stored non-temporally: written once, never re-read; keeps the bf16
// panels resident in L2 for the cross-tile re-reads.
// ---------------------------------------------------------------------------
__global__ __launch_bounds__(256)
void qjl_gemm(const unsigned short* __restrict__ Ap,
              const unsigned short* __restrict__ Bp,
              float* __restrict__ out) {
  __shared__ __attribute__((aligned(16))) unsigned short lA[TILE * KCH]; // 16 KB
  __shared__ __attribute__((aligned(16))) unsigned short lB[TILE * KCH]; // 16 KB

  const int h     = blockIdx.z;
  const int nBase = blockIdx.x * TILE;   // T (column) tile
  const int mBase = blockIdx.y * TILE;   // N (row) tile

  const unsigned short* gA = Ap + ((size_t)h * NROWS + mBase) * KTOT;
  const unsigned short* gB = Bp + ((size_t)h * TCOLS + nBase) * KTOT;

  const int tid   = threadIdx.x;
  const int wave  = tid >> 5;
  const int lane  = tid & 31;
  const int lm    = lane & 15;           // M (A) / N (B) within a 16-wide frag
  const int lh    = lane >> 4;           // lane-half selects K sub-block
  const int waveM = wave & 3;            // 4 waves along M
  const int waveN = wave >> 2;           // 2 waves along N

  v8f acc[2][4];
#pragma unroll
  for (int i = 0; i < 2; ++i)
#pragma unroll
    for (int j = 0; j < 4; ++j) acc[i][j] = (v8f)(0.0f);

  for (int kc = 0; kc < KTOT; kc += KCH) {
    // Prefetch next K panel of B into near caches while we work on this one.
    if (kc + KCH < KTOT)
      __builtin_prefetch(&gB[(size_t)(tid) * KTOT + kc + KCH], 0, 3);

    // Cooperative panel load: 128 rows x 64 bf16 each, b128 chunks.
    // 1024 16-byte chunks per panel -> 4 per thread per panel.
#pragma unroll
    for (int i = tid; i < TILE * KCH / 8; i += 256) {
      const int row = i >> 3, c8 = i & 7;
      ((uint4v*)lA)[i] =
          *(const uint4v*)&gA[(size_t)row * KTOT + kc + c8 * 8];
      ((uint4v*)lB)[i] =
          *(const uint4v*)&gB[(size_t)row * KTOT + kc + c8 * 8];
    }
    __syncthreads();

#pragma unroll
    for (int kk = 0; kk < KCH; kk += 32) {
      // A fragments (16x32 bf16): lane-half 0 holds K {0..7,16..23},
      // lane-half 1 holds K {8..15,24..31} -> two 16B ds reads per frag.
      union { uint4v u[2]; v16bf v; } fa[2];
#pragma unroll
      for (int mi = 0; mi < 2; ++mi) {
        const int row = waveM * 32 + mi * 16 + lm;
        fa[mi].u[0] = *(const uint4v*)&lA[row * KCH + kk + lh * 8];
        fa[mi].u[1] = *(const uint4v*)&lA[row * KCH + kk + 16 + lh * 8];
      }
      // B fragments (32x16 bf16): lane-half 0 -> K {0..7,16..23}, half 1 -> K {8..15,24..31}.
      union { uint4v u[2]; v16bf v; } fb[4];
#pragma unroll
      for (int ni = 0; ni < 4; ++ni) {
        const int col = waveN * 64 + ni * 16 + lm;
        fb[ni].u[0] = *(const uint4v*)&lB[col * KCH + kk + lh * 8];
        fb[ni].u[1] = *(const uint4v*)&lB[col * KCH + kk + 16 + lh * 8];
      }
#pragma unroll
      for (int mi = 0; mi < 2; ++mi)
#pragma unroll
        for (int ni = 0; ni < 4; ++ni)
          acc[mi][ni] = __builtin_amdgcn_wmma_f32_16x16x32_bf16(
              false, fa[mi].v, false, fb[ni].v,
              (short)0, acc[mi][ni], false, false);
    }
    __syncthreads();
  }

  // Store: C/D layout — VGPR v: lanes 0-15 = (M=v,   N=lane),
  //                              lanes 16-31 = (M=v+8, N=lane-16).
  // Non-temporal: streaming output, written once, never re-read.
#pragma unroll
  for (int mi = 0; mi < 2; ++mi) {
#pragma unroll
    for (int ni = 0; ni < 4; ++ni) {
      const int col = nBase + waveN * 64 + ni * 16 + lm;
#pragma unroll
      for (int v = 0; v < 8; ++v) {
        const int row = mBase + waveM * 32 + mi * 16 + lh * 8 + v;
        __builtin_nontemporal_store(
            acc[mi][ni][v], &out[((size_t)h * NROWS + row) * TCOLS + col]);
      }
    }
  }
}

// ---------------------------------------------------------------------------
extern "C" void kernel_launch(void* const* d_in, const int* in_sizes, int n_in,
                              void* d_out, int out_size, void* d_ws,
                              size_t ws_size, hipStream_t stream) {
  const float* q  = (const float*)d_in[0];  // (1,32,2048,128)
  const float* ko = (const float*)d_in[1];  // (1,32,2048,128)
  const float* kq = (const float*)d_in[2];  // (1,32,2048,128)
  const float* G  = (const float*)d_in[3];  // (128,64)
  float* out = (float*)d_out;               // (1,32,2048,2048)

  unsigned short* Ap = (unsigned short*)d_ws;                    // 25.2 MB
  unsigned short* Bp = Ap + (size_t)HEADS * NROWS * KTOT;        // 25.2 MB

  qjl_prep_q<<<dim3(HEADS * NROWS), dim3(64), 0, stream>>>(q, G, Ap);
  qjl_prep_k<<<dim3(HEADS * TCOLS), dim3(64), 0, stream>>>(ko, kq, G, Bp);
  qjl_gemm<<<dim3(TCOLS / TILE, NROWS / TILE, HEADS), dim3(256), 0, stream>>>(
      Ap, Bp, out);
}
